// AdvisorCrossAttentionAdapter_4681514352962
// MI455X (gfx1250) — matrix operand order repaired and checked by reading, not compiled, not run on hardware
//
#include <hip/hip_runtime.h>

// ---------------------------------------------------------------------------
// AdvisorCrossAttentionAdapter for MI455X (gfx1250, wave32, WMMA).
// B=8, S=2048, L=1536 (T=512 triplets), H=1024.
// ~137 GFLOP vs ~9us of HBM traffic at 23.3 TB/s -> compute bound -> all six
// GEMM stages run on v_wmma_f32_16x16x32_bf16 (8x FLOP/inst of f32 WMMA),
// f32 accumulate.  Projection GEMMs stage W tiles into LDS with the gfx1250
// async global->LDS path (double buffered, s_wait_asynccnt overlapped).
// ---------------------------------------------------------------------------

typedef __attribute__((ext_vector_type(16))) __bf16 v16bf;
typedef __attribute__((ext_vector_type(8)))  __bf16 v8bf;
typedef __attribute__((ext_vector_type(8)))  float  v8f;

#define H_ 1024
#define B_ 8
#define S_ 2048
#define L_ 1536
#define T_ 512

#if __has_builtin(__builtin_amdgcn_global_load_async_to_lds_b128)
#define ASYNC_LDS 1
#else
#define ASYNC_LDS 0
#endif

// The async builtin's params (from hipcc diagnostics) are pointers to a
// 16-byte int vector: global (AS1, printed "__device__") src, LDS (AS3) dst.
typedef int v4i_t __attribute__((vector_size(16)));
typedef __attribute__((address_space(1))) v4i_t g_v4i;
typedef __attribute__((address_space(3))) v4i_t l_v4i;

// Copy 32B (one B-fragment half-row) global -> LDS.  Async path uses two
// b128 async transfers (IOFFSET applies to both global and LDS addresses per
// ISA 08_async_tensor 4.4); fallback is a synchronous load + ds_store.
__device__ __forceinline__ void stage32(const __bf16* gsrc, __bf16* ldst) {
#if ASYNC_LDS
  g_v4i* gp = (g_v4i*)(uintptr_t)gsrc;
  l_v4i* lp = (l_v4i*)(uint32_t)(uintptr_t)ldst;
  __builtin_amdgcn_global_load_async_to_lds_b128(gp, lp, 0, 0);
  __builtin_amdgcn_global_load_async_to_lds_b128(gp, lp, 16, 0);
#else
  v8bf t0 = *(const v8bf*)gsrc;
  v8bf t1 = *(const v8bf*)(gsrc + 8);
  *(v8bf*)ldst = t0;
  *(v8bf*)(ldst + 8) = t1;
#endif
}

// Wait until at most KEEP async transfers remain outstanding (this wave).
template <int KEEP>
__device__ __forceinline__ void wait_async_keep() {
#if ASYNC_LDS
#if __has_builtin(__builtin_amdgcn_s_wait_asynccnt)
  __builtin_amdgcn_s_wait_asynccnt((unsigned short)KEEP);
#else
  asm volatile("s_wait_asynccnt %0" ::"i"(KEEP) : "memory");
#endif
#endif
}

__device__ __forceinline__ v8f vzero8() {
  v8f z;
#pragma unroll
  for (int i = 0; i < 8; ++i) z[i] = 0.0f;
  return z;
}

// ---------------------------------------------------------------------------
// Kernel 1: convert the four 1024x1024 f32 weight matrices to bf16 once.
// ---------------------------------------------------------------------------
__global__ void cvt_w_kernel(const float* __restrict__ wq, const float* __restrict__ wk,
                             const float* __restrict__ wv, const float* __restrict__ wo,
                             __bf16* __restrict__ dst) {
  size_t idx = (size_t)blockIdx.x * blockDim.x + threadIdx.x;
  if (idx >= (size_t)4 * H_ * H_) return;
  int w = (int)(idx >> 20);
  size_t r = idx & ((1u << 20) - 1);
  const float* s = (w == 0) ? wq : (w == 1) ? wk : (w == 2) ? wv : wo;
  dst[idx] = (__bf16)s[r];
}

// ---------------------------------------------------------------------------
// Kernel 2: generic NT GEMM  C[M,N] = A[M,K=1024] * W[N,K]^T
// Block = 128 threads = 4 waves; wave computes a 16x64 strip.
// W tiles (64 rows x 32 k, 4KB) are staged into LDS double buffers with
// async global->LDS copies: each thread moves one 32B chunk, the next tile's
// copy overlaps the current tile's 4x4 WMMA work (s_wait_asynccnt keeps the
// in-flight tile outstanding).  B-fragments then come from ds_load_b128.
// LDS row pitch = 40 bf16 (20 dwords) -> 16 lanes hit 16 distinct banks.
// WMMA bf16 operand layouts used here (contiguous k, same order in A and B):
//   A-frag: lane (m=lane&15, sel=lane>>4): K=[k0+sel*16..+15]
//   B-frag: lane (n=lane&15, khalf=lane>>4): K=[k0+khalf*16..+15]
//   C/D:    VGPR j -> row m0+j+(lane>>4)*8, col n0+(lane&15)
// ---------------------------------------------------------------------------
template <bool A_F32, bool OUT_F32>
__global__ __launch_bounds__(128) void gemm_nt_kernel(
    const void* __restrict__ Av, long lda, const __bf16* __restrict__ W,
    void* __restrict__ Cv, int Mrows) {
  __shared__ __bf16 wtile[2][64][40];

  const int lane = threadIdx.x & 31;
  const int wave = threadIdx.x >> 5;
  const int m0 = blockIdx.x * 64 + wave * 16;
  const int n0 = blockIdx.y * 64;
  const int mr  = lane & 15;   // A row in tile / B column in tile
  const int sel = lane >> 4;

  // cooperative staging role: 128 threads x 32B = 64x32 bf16 tile
  const int srow  = threadIdx.x >> 1;      // 0..63 (W row within tile)
  const int shalf = threadIdx.x & 1;       // which 16-element half

  const __bf16* wsrc_row = W + (size_t)(n0 + srow) * H_ + shalf * 16;

  v8f acc[4];
#pragma unroll
  for (int i = 0; i < 4; ++i) acc[i] = vzero8();

  // prologue: stage tile 0
  stage32(wsrc_row + 0, &wtile[0][srow][shalf * 16]);

  for (int kt = 0; kt < 32; ++kt) {
    const int cur = kt & 1;
    const int k0 = kt * 32;
    if (kt + 1 < 32) {
      stage32(wsrc_row + k0 + 32, &wtile[1 - cur][srow][shalf * 16]);
      wait_async_keep<2>();   // previous tile complete; next tile in flight
    } else {
      wait_async_keep<0>();
    }
    __syncthreads();          // tile[cur] visible to all waves

    // A fragment (per-wave, from global; f32 source converted to bf16)
    v16bf a;
    if (A_F32) {
      const float* A = (const float*)Av;
      const float* p = A + (size_t)(m0 + mr) * lda + k0 + sel * 16;
      v8f c0 = *(const v8f*)p;
      v8f c1 = *(const v8f*)(p + 8);
#pragma unroll
      for (int i = 0; i < 8; ++i) { a[i] = (__bf16)c0[i]; a[i + 8] = (__bf16)c1[i]; }
    } else {
      const __bf16* A = (const __bf16*)Av;
      const __bf16* p = A + (size_t)(m0 + mr) * lda + k0 + sel * 16;
      v8bf c0 = *(const v8bf*)p;
      v8bf c1 = *(const v8bf*)(p + 8);
#pragma unroll
      for (int i = 0; i < 8; ++i) { a[i] = c0[i]; a[i + 8] = c1[i]; }
    }

#pragma unroll
    for (int nn = 0; nn < 4; ++nn) {
      v16bf b = *(const v16bf*)&wtile[cur][nn * 16 + mr][sel * 16];
      acc[nn] = __builtin_amdgcn_wmma_f32_16x16x32_bf16(
          false, a, false, b, (short)0, acc[nn], false, false);
    }
    __syncthreads();          // all reads of tile[cur] done before re-staging
  }

  if (m0 >= Mrows) return;    // never taken (grids divide exactly)
#pragma unroll
  for (int nn = 0; nn < 4; ++nn) {
    const int col = n0 + nn * 16 + mr;
#pragma unroll
    for (int j = 0; j < 8; ++j) {
      const size_t row = (size_t)(m0 + sel * 8 + j);
      if (OUT_F32) ((float*)Cv)[row * H_ + col] = acc[nn][j];
      else ((__bf16*)Cv)[row * H_ + col] = (__bf16)acc[nn][j];
    }
  }
}

// ---------------------------------------------------------------------------
// Kernel 3: triplet logic-combine of V projections.
// Reads vproj_bf [B,T,3,H], ids (int64), writes v_final TRANSPOSED [B,H,T]
// so the attn@V WMMA B-operand (lane n = output channel, 16 contiguous t)
// is a single 32B load.
// ---------------------------------------------------------------------------
__global__ void combine_v_kernel(const __bf16* __restrict__ vp,
                                 const long long* __restrict__ ids,
                                 __bf16* __restrict__ vT) {
  size_t idx = (size_t)blockIdx.x * blockDim.x + threadIdx.x;
  if (idx >= (size_t)B_ * T_ * H_) return;
  const int h = (int)(idx & (H_ - 1));
  const size_t bt = idx >> 10;           // b*T + t
  const int t = (int)(bt & (T_ - 1));
  const int b = (int)(bt >> 9);
  const size_t base = bt * 3 * H_ + h;
  const float vr = (float)vp[base];
  const float v1 = (float)vp[base + H_];
  const float v2 = (float)vp[base + 2 * H_];
  const long long rel = ids[(size_t)b * L_ + 3 * t];
  float o;
  if (rel == 0)      o = fminf(v1, v2);
  else if (rel == 1) o = fmaxf(v1, v2);
  else if (rel == 2) o = -v1;
  else if (rel == 3) o = fmaxf(-v1, v2);
  else if (rel == 4) o = fabsf(v1 - v2);
  else               o = vr;             // rel >= 5: learned passthrough
  vT[((size_t)b * H_ + h) * T_ + t] = (__bf16)o;
}

// ---------------------------------------------------------------------------
// Kernel 4: fused attention.  Block = 4 waves; each wave owns 16 query rows
// and ALL T=512 keys: 32 score tiles (16x16 f32) live in 256 accumulator
// VGPRs (VGPR-MSB addressing).  In-register softmax via 16-lane xor-shuffle
// reductions matching the C/D layout; normalized probs staged to LDS in bf16
// and re-read in the A-fragment layout for the attn@V WMMA pass.
// ---------------------------------------------------------------------------
__global__ __launch_bounds__(128) void attn_kernel(
    const __bf16* __restrict__ qbf, const __bf16* __restrict__ kbf,
    const __bf16* __restrict__ vT, __bf16* __restrict__ ctx) {
  __shared__ __bf16 attn_lds[4][16][520];  // 520 = 512 + 8 pad (16B aligned)

  const int lane = threadIdx.x & 31;
  const int wave = threadIdx.x >> 5;
  const int b  = blockIdx.x >> 5;              // 32 blocks per batch (S/64)
  const int s0 = (blockIdx.x & 31) * 64 + wave * 16;
  const int mr  = lane & 15;
  const int sel = lane >> 4;

  // ---- Phase 1: scores = Q K^T ------------------------------------------
  v8f sc[32];
#pragma unroll
  for (int tt = 0; tt < 32; ++tt) sc[tt] = vzero8();

  const __bf16* qrow  = qbf + ((size_t)b * S_ + s0 + mr) * H_;
  const __bf16* kbase = kbf + (size_t)b * T_ * H_;

  for (int h = 0; h < H_; h += 32) {
    v8bf c0 = *(const v8bf*)(qrow + h + sel * 16);
    v8bf c1 = *(const v8bf*)(qrow + h + 8 + sel * 16);
    v16bf a;
#pragma unroll
    for (int i = 0; i < 8; ++i) { a[i] = c0[i]; a[i + 8] = c1[i]; }
#pragma unroll
    for (int tt = 0; tt < 32; ++tt) {
      const __bf16* kp = kbase + (size_t)(tt * 16 + mr) * H_ + h + sel * 16;
      v16bf bb = *(const v16bf*)kp;
      sc[tt] = __builtin_amdgcn_wmma_f32_16x16x32_bf16(
          false, a, false, bb, (short)0, sc[tt], false, false);
    }
  }

  // ---- Softmax over T (scale 1/sqrt(H) = 1/32) --------------------------
  float mx[8], sm[8];
#pragma unroll
  for (int j = 0; j < 8; ++j) mx[j] = -3.4e38f;
#pragma unroll
  for (int tt = 0; tt < 32; ++tt)
#pragma unroll
    for (int j = 0; j < 8; ++j) mx[j] = fmaxf(mx[j], sc[tt][j]);
#pragma unroll
  for (int off = 1; off < 16; off <<= 1)
#pragma unroll
    for (int j = 0; j < 8; ++j) mx[j] = fmaxf(mx[j], __shfl_xor(mx[j], off, 32));

#pragma unroll
  for (int j = 0; j < 8; ++j) sm[j] = 0.0f;
  const float isq = 0.03125f;  // 1/sqrt(1024)
#pragma unroll
  for (int tt = 0; tt < 32; ++tt)
#pragma unroll
    for (int j = 0; j < 8; ++j) {
      float e = __expf((sc[tt][j] - mx[j]) * isq);
      sc[tt][j] = e;
      sm[j] += e;
    }
#pragma unroll
  for (int off = 1; off < 16; off <<= 1)
#pragma unroll
    for (int j = 0; j < 8; ++j) sm[j] += __shfl_xor(sm[j], off, 32);
#pragma unroll
  for (int j = 0; j < 8; ++j) sm[j] = 1.0f / sm[j];

  // scatter normalized probs: element (row j+sel*8, col tt*16+mr)
#pragma unroll
  for (int tt = 0; tt < 32; ++tt)
#pragma unroll
    for (int j = 0; j < 8; ++j)
      attn_lds[wave][sel * 8 + j][tt * 16 + mr] = (__bf16)(sc[tt][j] * sm[j]);

  __syncthreads();

  // ---- Phase 2: context = attn @ V_final --------------------------------
  const __bf16* vb = vT + (size_t)b * H_ * T_;
  __bf16* crow = ctx + ((size_t)b * S_ + s0) * H_;
  for (int n0 = 0; n0 < H_; n0 += 16) {
    v8f acc = vzero8();
#pragma unroll
    for (int t0 = 0; t0 < T_; t0 += 32) {
      const __bf16* a0 = &attn_lds[wave][mr][t0 + sel * 16];
      v8bf c0 = *(const v8bf*)a0;
      v8bf c1 = *(const v8bf*)(a0 + 8);
      v16bf a;
#pragma unroll
      for (int i = 0; i < 8; ++i) { a[i] = c0[i]; a[i + 8] = c1[i]; }
      const __bf16* vp = vb + (size_t)(n0 + mr) * T_ + t0 + sel * 16;
      v16bf bb = *(const v16bf*)vp;
      acc = __builtin_amdgcn_wmma_f32_16x16x32_bf16(
          false, a, false, bb, (short)0, acc, false, false);
    }
#pragma unroll
    for (int j = 0; j < 8; ++j)
      crow[(size_t)(sel * 8 + j) * H_ + n0 + mr] = (__bf16)acc[j];
  }
}

// ---------------------------------------------------------------------------
// Host-side orchestration.  Workspace layout (bf16 elements):
//   [0,4M)    weights bf16 (Wq,Wk,Wv,Wo)
//   [4M,20M)  q_bf    [B,S,H]
//   [20M,24M) k_bf    [B,T,H]
//   [24M,36M) vproj   [B,T,3,H]
//   [36M,40M) vT      [B,H,T]
//   [40M,56M) ctx_bf  [B,S,H]          total 56M bf16 = 112 MB
// ---------------------------------------------------------------------------
extern "C" void kernel_launch(void* const* d_in, const int* in_sizes, int n_in,
                              void* d_out, int out_size, void* d_ws, size_t ws_size,
                              hipStream_t stream) {
  (void)in_sizes; (void)n_in; (void)out_size; (void)ws_size;
  const float*     hidden  = (const float*)d_in[0];
  const float*     advisor = (const float*)d_in[1];
  const long long* ids     = (const long long*)d_in[2];
  const float*     Wq      = (const float*)d_in[3];
  const float*     Wk      = (const float*)d_in[4];
  const float*     Wv      = (const float*)d_in[5];
  const float*     Wo      = (const float*)d_in[6];

  __bf16* wbf   = (__bf16*)d_ws;
  __bf16* Wq_bf = wbf + (size_t)0 * H_ * H_;
  __bf16* Wk_bf = wbf + (size_t)1 * H_ * H_;
  __bf16* Wv_bf = wbf + (size_t)2 * H_ * H_;
  __bf16* Wo_bf = wbf + (size_t)3 * H_ * H_;
  __bf16* q_bf  = wbf + (size_t)4 * H_ * H_;
  __bf16* k_bf  = q_bf  + (size_t)B_ * S_ * H_;
  __bf16* vp_bf = k_bf  + (size_t)B_ * T_ * H_;
  __bf16* vT_bf = vp_bf + (size_t)B_ * T_ * 3 * H_;
  __bf16* ctx_bf= vT_bf + (size_t)B_ * H_ * T_;

  cvt_w_kernel<<<(4u << 20) / 256, 256, 0, stream>>>(Wq, Wk, Wv, Wo, wbf);

  // Q = hidden @ Wq^T               (rows B*S, row stride H)
  gemm_nt_kernel<true, false><<<dim3((B_ * S_) / 64, H_ / 64), 128, 0, stream>>>(
      hidden, H_, Wq_bf, q_bf, B_ * S_);
  // Vproj = advisor @ Wv^T          (rows B*L, row stride H)
  gemm_nt_kernel<true, false><<<dim3((B_ * L_) / 64, H_ / 64), 128, 0, stream>>>(
      advisor, H_, Wv_bf, vp_bf, B_ * L_);
  // K = triplet-slot-0 @ Wk^T       (rows B*T, row stride 3H)
  gemm_nt_kernel<true, false><<<dim3((B_ * T_) / 64, H_ / 64), 128, 0, stream>>>(
      advisor, 3 * H_, Wk_bf, k_bf, B_ * T_);

  combine_v_kernel<<<(B_ * T_ * H_) / 256, 256, 0, stream>>>(vp_bf, ids, vT_bf);

  attn_kernel<<<B_ * (S_ / 64), 128, 0, stream>>>(q_bf, k_bf, vT_bf, ctx_bf);

  // out = context @ Wo^T, f32 output straight to d_out
  gemm_nt_kernel<false, true><<<dim3((B_ * S_) / 64, H_ / 64), 128, 0, stream>>>(
      ctx_bf, H_, Wo_bf, d_out, B_ * S_);
}
